// RouteChoiceModel_60705067762033
// MI455X (gfx1250) — compile-verified, matrix operand order checked by reading, not run on hardware
//
#include <hip/hip_runtime.h>
#include <math.h>

typedef _Float16 h16;
typedef __attribute__((ext_vector_type(16))) _Float16 v16h;
typedef __attribute__((ext_vector_type(8)))  float    v8f;
typedef __attribute__((ext_vector_type(4)))  unsigned int u32x4;
typedef __attribute__((ext_vector_type(8)))  int      i32x8;
typedef __attribute__((ext_vector_type(4)))  int      i32x4;

#define B_   8
#define T1_  128
#define T2_  8
#define C_   256
#define E_   8
#define NTOK (B_ * T1_)        // 1024 tokens
#define NROW (NTOK * T2_)      // 8192 adjacent rows

__device__ __forceinline__ float silu_f(float x) { return x / (1.0f + __expf(-x)); }

// ---------------------------------------------------------------------------
// Weight swizzle: fp32 row-major [K,N]  ->  f16 WMMA "B-fragment" layout:
//   [K/32][N/16][lane 0..31][16 halves], lane holds column n = nt*16 + (lane&15),
//   half i holds k = kt*32 + ((lane>=16)?8:0) + (i<8 ? i : i+8)
// (matches the 16-bit 16x32 per-lane K packing in CDNA5 ISA)
// ---------------------------------------------------------------------------
__global__ void k_swz(const float* __restrict__ W, h16* __restrict__ out,
                      int K, int N, long matStride)
{
    const float* w = W  + (long)blockIdx.z * matStride;
    h16*         o = out + (long)blockIdx.z * matStride;
    long idx   = (long)blockIdx.x * blockDim.x + threadIdx.x;
    long total = (long)K * N;
    if (idx >= total) return;
    int  i      = (int)(idx & 15);
    int  lane   = (int)((idx >> 4) & 31);
    long rest   = idx >> 9;
    int  ntiles = N >> 4;
    int  nt     = (int)(rest % ntiles);
    int  kt     = (int)(rest / ntiles);
    int  n = (nt << 4) + (lane & 15);
    int  k = (kt << 5) + ((lane >> 4) << 3) + (i < 8 ? i : i + 8);
    o[idx] = (h16)w[(long)k * N + n];
}

// ---------------------------------------------------------------------------
// adjc = concat(adj_feat, [sin,cos](pi*angle)@ang_w+ang_b, trp@trp_w+trp_b)
// f16, row-major [NROW, 768]
// ---------------------------------------------------------------------------
__global__ void k_adjc(const float* __restrict__ adj_feat,
                       const float* __restrict__ angle,
                       const float* __restrict__ transp,
                       const float* __restrict__ ang_w, const float* __restrict__ ang_b,
                       const float* __restrict__ trp_w, const float* __restrict__ trp_b,
                       h16* __restrict__ adjc)
{
    long idx = (long)blockIdx.x * blockDim.x + threadIdx.x;
    if (idx >= (long)NROW * 768) return;
    int row = (int)(idx / 768);
    int col = (int)(idx % 768);
    float v;
    if (col < C_) {
        v = adj_feat[(size_t)row * C_ + col];
    } else if (col < 2 * C_) {
        int c = col - C_;
        float a = angle[row] * 3.14159265358979323846f;
        v = __sinf(a) * ang_w[c] + __cosf(a) * ang_w[C_ + c] + ang_b[c];
    } else {
        int c = col - 2 * C_;
        v = transp[row] * trp_w[c] + trp_b[c];
    }
    adjc[idx] = (h16)v;
}

// cur = concat(traj_feat, progress@tp_w+tp_b)   f16 [NTOK, 512]
__global__ void k_cur(const float* __restrict__ traj_feat,
                      const float* __restrict__ tprog,
                      const float* __restrict__ tp_w, const float* __restrict__ tp_b,
                      h16* __restrict__ cur)
{
    int idx = blockIdx.x * blockDim.x + threadIdx.x;
    if (idx >= NTOK * 512) return;
    int tok = idx >> 9, col = idx & 511;
    float v = (col < C_) ? traj_feat[(size_t)tok * C_ + col]
                         : tprog[tok] * tp_w[col - C_] + tp_b[col - C_];
    cur[idx] = (h16)v;
}

// ---------------------------------------------------------------------------
// Gate: sigmoid(go), top-2 over go+expert_biases, normalized combine weights
// ---------------------------------------------------------------------------
__global__ void k_gate(const float* __restrict__ traj_feat,
                       const float* __restrict__ tprog,
                       const float* __restrict__ tp_w, const float* __restrict__ tp_b,
                       const float* __restrict__ gate_w, const float* __restrict__ gate_b,
                       const float* __restrict__ ebias,
                       float* __restrict__ cw, int* __restrict__ topk)
{
    int t = blockIdx.x * blockDim.x + threadIdx.x;
    if (t >= NTOK) return;
    float go[E_];
#pragma unroll
    for (int e = 0; e < E_; ++e) go[e] = gate_b[e];
    const float* xf = traj_feat + (size_t)t * C_;
    for (int i = 0; i < C_; ++i) {
        float x = xf[i];
        const float* g = gate_w + i * E_;
#pragma unroll
        for (int e = 0; e < E_; ++e) go[e] += x * g[e];
    }
    float pv = tprog[t];
    for (int i = 0; i < C_; ++i) {
        float x = pv * tp_w[i] + tp_b[i];
        const float* g = gate_w + (C_ + i) * E_;
#pragma unroll
        for (int e = 0; e < E_; ++e) go[e] += x * g[e];
    }
    float probs[E_], logit[E_];
#pragma unroll
    for (int e = 0; e < E_; ++e) {
        probs[e] = 1.0f / (1.0f + __expf(-go[e]));
        logit[e] = go[e] + ebias[e];
    }
    int i0 = 0; float m0 = logit[0];
#pragma unroll
    for (int e = 1; e < E_; ++e) if (logit[e] > m0) { m0 = logit[e]; i0 = e; }
    int i1 = -1; float m1 = -3.0e38f;
#pragma unroll
    for (int e = 0; e < E_; ++e) if (e != i0 && logit[e] > m1) { m1 = logit[e]; i1 = e; }
    float p0 = probs[i0], p1 = probs[i1], s = p0 + p1;
    p0 /= s; p1 /= s;
#pragma unroll
    for (int e = 0; e < E_; ++e) cw[t * E_ + e] = 0.0f;
    cw[t * E_ + i0] = p0;
    cw[t * E_ + i1] = p1;
    topk[t * 2]     = i0;
    topk[t * 2 + 1] = i1;
}

__global__ void k_maxvio(const int* __restrict__ topk,
                         const int* __restrict__ traj_len,
                         float* __restrict__ out_maxvio)
{
    __shared__ float counts[E_];
    int tid = threadIdx.x;
    if (tid < E_) counts[tid] = 0.0f;
    __syncthreads();
    for (int t = tid; t < NTOK; t += blockDim.x) {
        int b = t >> 7, t1 = t & 127;
        if (t1 < traj_len[b]) {
            atomicAdd(&counts[topk[t * 2]], 1.0f);
            atomicAdd(&counts[topk[t * 2 + 1]], 1.0f);
        }
    }
    __syncthreads();
    if (tid == 0) {
        float mx = counts[0], mean = 0.0f;
#pragma unroll
        for (int e = 0; e < E_; ++e) { mx = fmaxf(mx, counts[e]); mean += counts[e]; }
        mean *= (1.0f / E_);
        *out_maxvio = (mx - mean) / (mean + 1e-5f);
    }
}

// ---------------------------------------------------------------------------
// GEMM tiling (shared by both WMMA kernels):
//   block = 128 threads = 4 wave32s arranged 2(M) x 2(N)
//   block tile = 64 rows x 64 cols, wave tile = 32x32 (2x2 16x16 WMMA D tiles)
//   K-loop step = 32; every A/B fragment feeds two v_wmma_f32_16x16x32_f16
//   A tile staged into LDS (TDM tensor_load_to_lds in the SwiGLU kernel,
//   manual ds_store_b128 in the down-projection kernel); B fragments read
//   straight from the pre-swizzled weight layout (2x global_load_b128,
//   wave-coalesced, L2-resident across all row blocks).
// ---------------------------------------------------------------------------
__device__ __forceinline__ v16h frag_from_lds(const _Float16* p0, const _Float16* p1)
{
    v16h r;
    ((uint4*)&r)[0] = *(const uint4*)p0;
    ((uint4*)&r)[1] = *(const uint4*)p1;
    return r;
}
__device__ __forceinline__ v16h frag_from_global(const h16* p)
{
    v16h r;
    ((uint4*)&r)[0] = *(const uint4*)p;
    ((uint4*)&r)[1] = *(const uint4*)(p + 8);
    return r;
}

// Issue one TDM descriptor: 64 rows x 32 halves tile at X+(m0*K+k0)*2 -> LDS 0
__device__ __forceinline__ void tdm_stage_tile(const h16* X, int m0, int k0,
                                               int M, int K)
{
    unsigned long long ga =
        (unsigned long long)(size_t)X + ((unsigned long long)m0 * (unsigned)K + (unsigned)k0) * 2ull;
    u32x4 g0;
    g0[0] = 1u;                                   // count=1, user descriptor
    g0[1] = 0u;                                   // lds_addr = 0 (ldsA base)
    g0[2] = (unsigned)(ga & 0xffffffffu);         // global_addr[31:0]
    g0[3] = (unsigned)((ga >> 32) & 0x01ffffffu)  // global_addr[56:32]
          | (2u << 30);                           // type = 2 ("image")
    i32x8 g1;
    g1[0] = 1 << 16;                              // data_size = 1 -> 2 bytes
    g1[1] = (K & 0xffff) << 16;                   // tensor_dim0[15:0]
    g1[2] = ((unsigned)K >> 16) | ((M & 0xffff) << 16);          // dim0 hi | dim1 lo
    g1[3] = (((unsigned)M >> 16) & 0xffff) | (32 << 16);         // dim1 hi | tile_dim0=32
    g1[4] = 64;                                   // tile_dim1 = 64 rows (tile_dim2=0)
    g1[5] = K;                                    // tensor_dim0_stride[31:0]
    g1[6] = 0;                                    // stride hi | dim1_stride lo
    g1[7] = 0;
    i32x4 z4 = {0, 0, 0, 0};
    i32x8 z8 = {0, 0, 0, 0, 0, 0, 0, 0};
    __builtin_amdgcn_tensor_load_to_lds(g0, g1, z4, z4, z8, 0);
    __builtin_amdgcn_s_wait_tensorcnt(0);
}

// H = silu(X@W1 + b1) * (X@W2 + b2)   -> f16 [M,N]
__global__ void __launch_bounds__(128)
k_swiglu_hidden(const h16* __restrict__ X,
                const h16* __restrict__ W1, const float* __restrict__ b1,
                const h16* __restrict__ W2, const float* __restrict__ b2,
                h16* __restrict__ H, int M, int K, int N)
{
    __shared__ __align__(16) _Float16 ldsA[64 * 32];
    const int tid   = threadIdx.x;
    const int lane  = tid & 31;
    const int wave  = tid >> 5;
    const int waveM = wave >> 1;
    const int waveN = wave & 1;
    const int m0 = blockIdx.y * 64;
    const int n0 = blockIdx.x * 64;
    const int ntiles = N >> 4;

    // Dynamically dead store (M is always > 0): keeps global-opt from
    // classifying ldsA as never-written and folding its loads to undef
    // (the real writer is the TDM engine, invisible to the compiler).
    if (M == -1) ldsA[tid] = (h16)0.0f;

    v8f acc1[2][2] = {};   // [mi][ni] W1 path
    v8f acc2[2][2] = {};   // [mi][ni] W2 path

    const int kLo = (lane >> 4) << 3;

    for (int k0 = 0; k0 < K; k0 += 32) {
        __syncthreads();
        if (wave == 0) tdm_stage_tile(X, m0, k0, M, K);   // TDM -> LDS, waits TENSORcnt
        __syncthreads();
        // Escape ldsA's address so the optimizer must assume the TDM (modeled
        // by this asm's memory clobber) rewrote the LDS tile this iteration.
        asm volatile("" :: "v"(ldsA) : "memory");

        v16h a[2];
#pragma unroll
        for (int mi = 0; mi < 2; ++mi) {
            const int aRow = (waveM << 5) + (mi << 4) + (lane & 15);
            a[mi] = frag_from_lds(&ldsA[aRow * 32 + kLo], &ldsA[aRow * 32 + kLo + 16]);
        }
        const int kt  = k0 >> 5;
        const int ntA = (n0 >> 4) + (waveN << 1);
#pragma unroll
        for (int ni = 0; ni < 2; ++ni) {
            v16h b1f = frag_from_global(W1 + (((size_t)kt * ntiles + ntA + ni) * 32 + lane) * 16);
#pragma unroll
            for (int mi = 0; mi < 2; ++mi)
                acc1[mi][ni] = __builtin_amdgcn_wmma_f32_16x16x32_f16(
                    false, a[mi], false, b1f, (short)0, acc1[mi][ni], false, false);
            v16h b2f = frag_from_global(W2 + (((size_t)kt * ntiles + ntA + ni) * 32 + lane) * 16);
#pragma unroll
            for (int mi = 0; mi < 2; ++mi)
                acc2[mi][ni] = __builtin_amdgcn_wmma_f32_16x16x32_f16(
                    false, a[mi], false, b2f, (short)0, acc2[mi][ni], false, false);
        }
    }

    // C/D layout: lane holds column n = (lane&15); VGPR r holds row r (+8 for lanes 16-31)
#pragma unroll
    for (int mi = 0; mi < 2; ++mi) {
        const int mBase = m0 + (waveM << 5) + (mi << 4) + ((lane >> 4) << 3);
#pragma unroll
        for (int ni = 0; ni < 2; ++ni) {
            const int n = n0 + (waveN << 5) + (ni << 4) + (lane & 15);
            const float bb1 = b1[n], bb2 = b2[n];
#pragma unroll
            for (int r = 0; r < 8; ++r) {
                float h = silu_f(acc1[mi][ni][r] + bb1) * (acc2[mi][ni][r] + bb2);
                H[(size_t)(mBase + r) * N + n] = (h16)h;
            }
        }
    }
}

// Out = (accum ? Out : 0) + scale_row * (X@W + bias)   -> f32 [M,N]
// scale_row[m] = cw ? cw[(m>>3)*8 + expert] : 1.0
__global__ void __launch_bounds__(128)
k_linear_out(const h16* __restrict__ X,
             const h16* __restrict__ W, const float* __restrict__ bias,
             float* __restrict__ Out, int M, int K, int N,
             const float* __restrict__ cw, int expert, int accumulate)
{
    (void)M;
    __shared__ __align__(16) _Float16 ldsA[64 * 32];
    const int tid   = threadIdx.x;
    const int lane  = tid & 31;
    const int wave  = tid >> 5;
    const int waveM = wave >> 1;
    const int waveN = wave & 1;
    const int m0 = blockIdx.y * 64;
    const int n0 = blockIdx.x * 64;
    const int ntiles = N >> 4;

    v8f acc[2][2] = {};
    const int kLo = (lane >> 4) << 3;

    for (int k0 = 0; k0 < K; k0 += 32) {
        __syncthreads();
#pragma unroll
        for (int c = 0; c < 2; ++c) {                 // 256 16B chunks, 128 threads
            int id  = tid + (c << 7);
            int row = id >> 2, seg = (id & 3) << 3;
            *(uint4*)&ldsA[row * 32 + seg] =
                *(const uint4*)&X[(size_t)(m0 + row) * K + k0 + seg];
        }
        __syncthreads();

        v16h a[2];
#pragma unroll
        for (int mi = 0; mi < 2; ++mi) {
            const int aRow = (waveM << 5) + (mi << 4) + (lane & 15);
            a[mi] = frag_from_lds(&ldsA[aRow * 32 + kLo], &ldsA[aRow * 32 + kLo + 16]);
        }
        const int kt  = k0 >> 5;
        const int ntA = (n0 >> 4) + (waveN << 1);
#pragma unroll
        for (int ni = 0; ni < 2; ++ni) {
            v16h bf = frag_from_global(W + (((size_t)kt * ntiles + ntA + ni) * 32 + lane) * 16);
#pragma unroll
            for (int mi = 0; mi < 2; ++mi)
                acc[mi][ni] = __builtin_amdgcn_wmma_f32_16x16x32_f16(
                    false, a[mi], false, bf, (short)0, acc[mi][ni], false, false);
        }
    }

#pragma unroll
    for (int mi = 0; mi < 2; ++mi) {
        const int mBase = m0 + (waveM << 5) + (mi << 4) + ((lane >> 4) << 3);
#pragma unroll
        for (int ni = 0; ni < 2; ++ni) {
            const int n = n0 + (waveN << 5) + (ni << 4) + (lane & 15);
            const float bb = bias[n];
#pragma unroll
            for (int r = 0; r < 8; ++r) {
                int m = mBase + r;
                float s = cw ? cw[(size_t)(m >> 3) * E_ + expert] : 1.0f;
                float v = s * (acc[mi][ni][r] + bb);
                size_t o = (size_t)m * N + n;
                if (accumulate) Out[o] += v;
                else            Out[o]  = v;
            }
        }
    }
}

// ---------------------------------------------------------------------------
// Attention over the T2=8 choices per token; z = concat(sel_emb, unsel_emb) f16
// ---------------------------------------------------------------------------
__global__ void __launch_bounds__(256)
k_attn(const float* __restrict__ q, const float* __restrict__ y,
       const unsigned char* __restrict__ selmask,
       const unsigned char* __restrict__ unselmask,
       h16* __restrict__ z)
{
    __shared__ float s_p[T2_];
    __shared__ float s_sel[T2_];
    __shared__ float s_uns[T2_];
    const int tok  = blockIdx.x;
    const int tid  = threadIdx.x;
    const int lane = tid & 31;
    const int wave = tid >> 5;            // 8 waves == 8 choices

    const float* qrow = q + (size_t)tok * C_;
    const float* krow = y + ((size_t)tok * T2_ + wave) * (2 * C_);
    float part = 0.0f;
    for (int c = lane; c < C_; c += 32) part += qrow[c] * krow[c];
#pragma unroll
    for (int off = 16; off; off >>= 1) part += __shfl_xor(part, off, 32);
    if (lane == 0) {
        bool uns = unselmask[tok * T2_ + wave] != 0;
        s_p[wave]   = uns ? part * (1.0f / 16.0f) : -1e9f;   // 1/sqrt(256)
        s_uns[wave] = uns ? 1.0f : 0.0f;
        s_sel[wave] = (selmask[tok * T2_ + wave] != 0) ? 1.0f : 0.0f;
    }
    __syncthreads();

    if (tid == 0) {
        float mx = s_p[0];
#pragma unroll
        for (int a = 1; a < T2_; ++a) mx = fmaxf(mx, s_p[a]);
        float e[T2_], sum = 0.0f, any = 0.0f;
#pragma unroll
        for (int a = 0; a < T2_; ++a) { e[a] = __expf(s_p[a] - mx); sum += e[a]; any += s_uns[a]; }
        float inv = (any > 0.0f) ? 1.0f / sum : 0.0f;
#pragma unroll
        for (int a = 0; a < T2_; ++a) s_p[a] = e[a] * inv;
    }
    __syncthreads();

    for (int c = tid; c < C_; c += blockDim.x) {
        float se = 0.0f, ue = 0.0f;
#pragma unroll
        for (int a = 0; a < T2_; ++a) {
            float v = y[((size_t)tok * T2_ + a) * (2 * C_) + C_ + c];
            se += s_sel[a] * v;
            ue += s_p[a] * v;
        }
        z[(size_t)tok * (2 * C_) + c]      = (h16)se;
        z[(size_t)tok * (2 * C_) + C_ + c] = (h16)ue;
    }
}

// ---------------------------------------------------------------------------
extern "C" void kernel_launch(void* const* d_in, const int* in_sizes, int n_in,
                              void* d_out, int out_size, void* d_ws, size_t ws_size,
                              hipStream_t stream)
{
    (void)in_sizes; (void)n_in; (void)out_size; (void)ws_size;
    const float* traj_feat = (const float*)d_in[0];
    const int*   traj_len  = (const int*)  d_in[1];
    const float* adj_feat  = (const float*)d_in[2];
    const float* tprog     = (const float*)d_in[3];
    const float* angle     = (const float*)d_in[4];
    const float* transp    = (const float*)d_in[5];
    const unsigned char* selmask   = (const unsigned char*)d_in[6];
    const unsigned char* unselmask = (const unsigned char*)d_in[7];
    const float* tp_w  = (const float*)d_in[8],  *tp_b  = (const float*)d_in[9];
    const float* ang_w = (const float*)d_in[10], *ang_b = (const float*)d_in[11];
    const float* trp_w = (const float*)d_in[12], *trp_b = (const float*)d_in[13];
    const float* q_w   = (const float*)d_in[14], *q_b   = (const float*)d_in[15];
    const float* gate_w= (const float*)d_in[16], *gate_b= (const float*)d_in[17];
    const float* ebias = (const float*)d_in[18];
    const float* ew1 = (const float*)d_in[19], *eb1 = (const float*)d_in[20];
    const float* ew2 = (const float*)d_in[21], *eb2 = (const float*)d_in[22];
    const float* ew3 = (const float*)d_in[23], *eb3 = (const float*)d_in[24];
    const float* sw1 = (const float*)d_in[25], *sb1 = (const float*)d_in[26];
    const float* sw2 = (const float*)d_in[27], *sb2 = (const float*)d_in[28];
    const float* sw3 = (const float*)d_in[29], *sb3 = (const float*)d_in[30];
    const float* fw1 = (const float*)d_in[31], *fb1 = (const float*)d_in[32];
    const float* fw2 = (const float*)d_in[33], *fb2 = (const float*)d_in[34];
    const float* fw3 = (const float*)d_in[35], *fb3 = (const float*)d_in[36];

    char* ws = (char*)d_ws;
    size_t off = 0;
    auto alloc = [&](size_t bytes) { size_t o = off; off += (bytes + 255) & ~(size_t)255; return o; };

    h16*   adjc_h = (h16*)  (ws + alloc((size_t)NROW * 768 * 2));
    h16*   cur_h  = (h16*)  (ws + alloc((size_t)NTOK * 512 * 2));
    h16*   z_h    = (h16*)  (ws + alloc((size_t)NTOK * 512 * 2));
    h16*   H_h    = (h16*)  (ws + alloc((size_t)NROW * 1024 * 2));
    float* y_f    = (float*)(ws + alloc((size_t)NROW * 512 * 4));
    float* q_f    = (float*)(ws + alloc((size_t)NTOK * 256 * 4));
    float* cw_f   = (float*)(ws + alloc((size_t)NTOK * E_ * 4));
    int*   topk_i = (int*)  (ws + alloc((size_t)NTOK * 2 * 4));
    h16*   ew1_h  = (h16*)  (ws + alloc((size_t)E_ * 768 * 1024 * 2));
    h16*   ew2_h  = (h16*)  (ws + alloc((size_t)E_ * 768 * 1024 * 2));
    h16*   ew3_h  = (h16*)  (ws + alloc((size_t)E_ * 1024 * 512 * 2));
    h16*   sw1_h  = (h16*)  (ws + alloc((size_t)768 * 1024 * 2));
    h16*   sw2_h  = (h16*)  (ws + alloc((size_t)768 * 1024 * 2));
    h16*   sw3_h  = (h16*)  (ws + alloc((size_t)1024 * 512 * 2));
    h16*   fw1_h  = (h16*)  (ws + alloc((size_t)512 * 1024 * 2));
    h16*   fw2_h  = (h16*)  (ws + alloc((size_t)512 * 1024 * 2));
    h16*   fw3_h  = (h16*)  (ws + alloc((size_t)1024 * 256 * 2));
    h16*   qw_h   = (h16*)  (ws + alloc((size_t)512 * 256 * 2));

    auto swz = [&](const float* W, h16* o, int K, int N, int nMat) {
        long per = (long)K * N;
        dim3 g((unsigned)((per + 255) / 256), 1, (unsigned)nMat);
        k_swz<<<g, dim3(256), 0, stream>>>(W, o, K, N, per);
    };
    dim3 blk(128);
    auto kA = [&](const h16* X, const h16* W1h, const float* bb1,
                  const h16* W2h, const float* bb2, h16* H, int M, int K, int N) {
        k_swiglu_hidden<<<dim3(N / 64, M / 64), blk, 0, stream>>>(X, W1h, bb1, W2h, bb2, H, M, K, N);
    };
    auto kB = [&](const h16* X, const h16* Wh, const float* bb, float* O,
                  int M, int K, int N, const float* cwp, int ex, int accum) {
        k_linear_out<<<dim3(N / 64, M / 64), blk, 0, stream>>>(X, Wh, bb, O, M, K, N, cwp, ex, accum);
    };

    // --- prep: activations + weight swizzles ---
    k_adjc<<<dim3((NROW * 768 + 255) / 256), dim3(256), 0, stream>>>(
        adj_feat, angle, transp, ang_w, ang_b, trp_w, trp_b, adjc_h);
    k_cur<<<dim3((NTOK * 512 + 255) / 256), dim3(256), 0, stream>>>(
        traj_feat, tprog, tp_w, tp_b, cur_h);
    swz(ew1, ew1_h, 768, 1024, E_);
    swz(ew2, ew2_h, 768, 1024, E_);
    swz(ew3, ew3_h, 1024, 512, E_);
    swz(sw1, sw1_h, 768, 1024, 1);
    swz(sw2, sw2_h, 768, 1024, 1);
    swz(sw3, sw3_h, 1024, 512, 1);
    swz(fw1, fw1_h, 512, 1024, 1);
    swz(fw2, fw2_h, 512, 1024, 1);
    swz(fw3, fw3_h, 1024, 256, 1);
    swz(q_w, qw_h, 512, 256, 1);

    // --- gate / top-k / maxvio ---
    k_gate<<<dim3((NTOK + 255) / 256), dim3(256), 0, stream>>>(
        traj_feat, tprog, tp_w, tp_b, gate_w, gate_b, ebias, cw_f, topk_i);
    k_maxvio<<<dim3(1), dim3(256), 0, stream>>>(topk_i, traj_len, (float*)d_out + (size_t)NTOK * C_);

    // --- shared expert (writes y), then routed experts (accumulate cw_e * ye) ---
    kA(adjc_h, sw1_h, sb1, sw2_h, sb2, H_h, NROW, 768, 1024);
    kB(H_h, sw3_h, sb3, y_f, NROW, 1024, 512, nullptr, 0, 0);
    for (int e = 0; e < E_; ++e) {
        kA(adjc_h, ew1_h + (size_t)e * 768 * 1024, eb1 + e * 1024,
                   ew2_h + (size_t)e * 768 * 1024, eb2 + e * 1024, H_h, NROW, 768, 1024);
        kB(H_h, ew3_h + (size_t)e * 1024 * 512, eb3 + e * 512, y_f,
           NROW, 1024, 512, cw_f, e, 1);
    }

    // --- q projection, attention, final SwiGLU head ---
    kB(cur_h, qw_h, q_b, q_f, NTOK, 512, 256, nullptr, 0, 0);
    k_attn<<<dim3(NTOK), dim3(256), 0, stream>>>(q_f, y_f, selmask, unselmask, z_h);
    kA(z_h, fw1_h, fb1, fw2_h, fb2, H_h, NTOK, 512, 1024);
    kB(H_h, fw3_h, fb3, (float*)d_out, NTOK, 1024, 256, nullptr, 0, 0);
}
